// UltimateNeuralNetwork_50964081934818
// MI455X (gfx1250) — compile-verified, matrix operand order checked by reading, not run on hardware
//
#include <hip/hip_runtime.h>
#include <hip/hip_bf16.h>
#include <math.h>

#define DIM 2048
#define NDEPTH 9

typedef __attribute__((ext_vector_type(16))) _Float16 v16h;
typedef __attribute__((ext_vector_type(8)))  float    v8f;

union Frag {
  v16h h;
  uint4 u[2];
};

union H4 {
  _Float16 h[4];
  uint2 u;
};

// Async global -> LDS copy, 16 bytes per lane (CDNA5, tracked by ASYNCcnt).
__device__ __forceinline__ void async_load_b128(unsigned int lds_off, const void* gptr) {
  asm volatile("global_load_async_to_lds_b128 %0, %1, off"
               :: "v"(lds_off), "v"((unsigned long long)(uintptr_t)gptr)
               : "memory");
}
__device__ __forceinline__ void wait_async0() {
  asm volatile("s_wait_asynccnt 0x0" ::: "memory");
}

// prob[i] = (1/DIM) * prod_{d<9, k<3} cos^2(uw[d, i, k])
__global__ void prob_kernel(const float* __restrict__ uw, float* __restrict__ prob) {
  int i = blockIdx.x * blockDim.x + threadIdx.x;
  if (i >= DIM) return;
  float p = 1.0f / (float)DIM;
  for (int d = 0; d < NDEPTH; ++d) {
    const float* a = uw + (size_t)d * DIM * DIM + (size_t)i * DIM;
    for (int k = 0; k < 3; ++k) {
      float c = cosf(a[k]);
      p *= c * c;
    }
  }
  prob[i] = p;
}

// dst[n*DIM + k] = (f16) src[k*DIM + n]   (transpose + convert)
__global__ void transpose_to_f16(const float* __restrict__ src, _Float16* __restrict__ dst) {
  __shared__ float tile[32][33];
  int bx = blockIdx.x * 32, by = blockIdx.y * 32;
  int tx = threadIdx.x, ty = threadIdx.y;  // 32 x 8
  for (int i = 0; i < 32; i += 8)
    tile[ty + i][tx] = src[(size_t)(by + ty + i) * DIM + bx + tx];
  __syncthreads();
  for (int i = 0; i < 32; i += 8)
    dst[(size_t)(bx + ty + i) * DIM + by + tx] = (_Float16)tile[tx][ty + i];
}

// C[m][n] = act( sum_k A[m][k]*B[n][k] + bias[n] (+ extra[n]) )
// A: M x K row-major (f32 or f16); B: N x K row-major (f32 or f16)
// f16 operands go global->LDS via async DMA; f32 operands are staged through
// VGPRs with f32->f16 conversion.
// ACT: 0 = relu, 1 = tanh (adds extra[n])
template <bool A16, bool B16, int ACT, bool OUT16>
__global__ __launch_bounds__(256) void gemm_wmma(
    const void* __restrict__ Ap, const void* __restrict__ Bp,
    const float* __restrict__ bias, const float* __restrict__ extra,
    void* __restrict__ Cp) {
  constexpr int BM = 128, BN = 128, BK = 32;
  constexpr bool ANY_ASYNC = A16 || B16;
  __shared__ _Float16 As[2][BM * BK];
  __shared__ _Float16 Bs[2][BN * BK];

  const int tid = threadIdx.x;
  const int m0 = blockIdx.y * BM;
  const int n0 = blockIdx.x * BN;

  const int wid = tid >> 5;
  const int lane = tid & 31;
  const int wm = (wid >> 1) * 32;   // 4 waves along M
  const int wn = (wid & 1) * 64;    // 2 waves along N
  const int lr = lane & 15;
  const int kq = (lane & 16) ? 8 : 0;  // K half-select per documented layout

  const float*    Af = (const float*)Ap;
  const _Float16* Ah = (const _Float16*)Ap;
  const float*    Bf = (const float*)Bp;
  const _Float16* Bh = (const _Float16*)Bp;

  v8f zero = {};
  v8f acc[2][4];
  for (int mt = 0; mt < 2; ++mt)
    for (int nt = 0; nt < 4; ++nt) acc[mt][nt] = zero;

  float4 aF[4], bF[4];

  // Async DMA for f16 operands: 2 x b128 per thread per operand per stage.
  auto asyncLoads = [&](int kk, int buf) {
    if constexpr (A16) {
      for (int i = 0; i < 2; ++i) {
        int idx = tid + i * 256;  // 512 x (8 halfs)
        int row = idx >> 2, c8 = idx & 3;
        const _Float16* g = Ah + (size_t)(m0 + row) * DIM + kk + c8 * 8;
        unsigned int l = (unsigned int)(uintptr_t)&As[buf][row * BK + c8 * 8];
        async_load_b128(l, g);
      }
    }
    if constexpr (B16) {
      for (int i = 0; i < 2; ++i) {
        int idx = tid + i * 256;
        int row = idx >> 2, c8 = idx & 3;
        const _Float16* g = Bh + (size_t)(n0 + row) * DIM + kk + c8 * 8;
        unsigned int l = (unsigned int)(uintptr_t)&Bs[buf][row * BK + c8 * 8];
        async_load_b128(l, g);
      }
    }
  };

  // Staged path for f32 operands (needs f32->f16 convert).
  auto loadGlobalF32 = [&](int kk) {
    if constexpr (!A16) {
      for (int i = 0; i < 4; ++i) {
        int idx = tid + i * 256;  // 1024 x (4 floats)
        int row = idx >> 3, c4 = idx & 7;
        aF[i] = *(const float4*)(Af + (size_t)(m0 + row) * DIM + kk + c4 * 4);
      }
    }
    if constexpr (!B16) {
      for (int i = 0; i < 4; ++i) {
        int idx = tid + i * 256;
        int row = idx >> 3, c4 = idx & 7;
        bF[i] = *(const float4*)(Bf + (size_t)(n0 + row) * DIM + kk + c4 * 4);
      }
    }
  };

  auto storeF32 = [&](int buf) {
    if constexpr (!A16) {
      for (int i = 0; i < 4; ++i) {
        int idx = tid + i * 256;
        int row = idx >> 3, c4 = idx & 7;
        H4 t;
        t.h[0] = (_Float16)aF[i].x; t.h[1] = (_Float16)aF[i].y;
        t.h[2] = (_Float16)aF[i].z; t.h[3] = (_Float16)aF[i].w;
        *(uint2*)&As[buf][row * BK + c4 * 4] = t.u;
      }
    }
    if constexpr (!B16) {
      for (int i = 0; i < 4; ++i) {
        int idx = tid + i * 256;
        int row = idx >> 3, c4 = idx & 7;
        H4 t;
        t.h[0] = (_Float16)bF[i].x; t.h[1] = (_Float16)bF[i].y;
        t.h[2] = (_Float16)bF[i].z; t.h[3] = (_Float16)bF[i].w;
        *(uint2*)&Bs[buf][row * BK + c4 * 4] = t.u;
      }
    }
  };

  auto compute = [&](int buf) {
    Frag a[2], b[4];
    for (int mt = 0; mt < 2; ++mt) {
      const _Float16* p = &As[buf][(wm + mt * 16 + lr) * BK];
      a[mt].u[0] = *(const uint4*)(p + kq);
      a[mt].u[1] = *(const uint4*)(p + kq + 16);
    }
    for (int nt = 0; nt < 4; ++nt) {
      const _Float16* p = &Bs[buf][(wn + nt * 16 + lr) * BK];
      b[nt].u[0] = *(const uint4*)(p + kq);
      b[nt].u[1] = *(const uint4*)(p + kq + 16);
    }
    for (int mt = 0; mt < 2; ++mt)
      for (int nt = 0; nt < 4; ++nt)
        acc[mt][nt] = __builtin_amdgcn_wmma_f32_16x16x32_f16(
            false, a[mt].h, false, b[nt].h, (short)0, acc[mt][nt], false, false);
  };

  constexpr int NSTEP = DIM / BK;  // 64
  asyncLoads(0, 0);
  loadGlobalF32(0);
  storeF32(0);
  if constexpr (ANY_ASYNC) wait_async0();
  __syncthreads();
  for (int s = 0; s < NSTEP; ++s) {
    int cur = s & 1;
    if (s + 1 < NSTEP) {
      asyncLoads((s + 1) * BK, cur ^ 1);  // DMA next stage while we compute
      loadGlobalF32((s + 1) * BK);
    }
    compute(cur);
    if (s + 1 < NSTEP) storeF32(cur ^ 1);
    if constexpr (ANY_ASYNC) wait_async0();
    __syncthreads();
  }

  // Epilogue. Acc layout: row = vgpr + (lane>=16 ? 8 : 0), col = lane%16.
  float* Cf = (float*)Cp;
  _Float16* Ch = (_Float16*)Cp;
  const int rbase = m0 + wm + ((lane >> 4) << 3);
  for (int nt = 0; nt < 4; ++nt) {
    int col = n0 + wn + nt * 16 + lr;
    float bv = bias[col];
    if constexpr (ACT == 1) bv += extra[col];
    for (int mt = 0; mt < 2; ++mt) {
      for (int v = 0; v < 8; ++v) {
        int row = rbase + mt * 16 + v;
        float r = acc[mt][nt][v] + bv;
        if constexpr (ACT == 1) r = tanhf(r);
        else r = fmaxf(r, 0.f);
        if constexpr (OUT16) Ch[(size_t)row * DIM + col] = (_Float16)r;
        else Cf[(size_t)row * DIM + col] = r;
      }
    }
  }
}

extern "C" void kernel_launch(void* const* d_in, const int* in_sizes, int n_in,
                              void* d_out, int out_size, void* d_ws, size_t ws_size,
                              hipStream_t stream) {
  const float* x   = (const float*)d_in[0];
  const float* w0  = (const float*)d_in[1];
  const float* b0  = (const float*)d_in[2];
  const float* w1  = (const float*)d_in[3];
  const float* b1  = (const float*)d_in[4];
  const float* uw1 = (const float*)d_in[5];
  const float* cw1 = (const float*)d_in[6];
  const float* cb1 = (const float*)d_in[7];
  const float* w2  = (const float*)d_in[8];
  const float* b2  = (const float*)d_in[9];
  const float* uw2 = (const float*)d_in[10];
  const float* cw2 = (const float*)d_in[11];
  const float* cb2 = (const float*)d_in[12];
  const float* wf  = (const float*)d_in[13];
  const float* bf  = (const float*)d_in[14];
  float* out = (float*)d_out;

  // Workspace: 2 f16 activation ping-pong + 2 f16 transposed cw + 2 prob vecs
  _Float16* Ha  = (_Float16*)d_ws;
  _Float16* Hb  = Ha + (size_t)DIM * DIM;
  _Float16* C1T = Hb + (size_t)DIM * DIM;
  _Float16* C2T = C1T + (size_t)DIM * DIM;
  float* prob1 = (float*)(C2T + (size_t)DIM * DIM);
  float* prob2 = prob1 + DIM;

  dim3 tb(32, 8), tg(DIM / 32, DIM / 32);
  transpose_to_f16<<<tg, tb, 0, stream>>>(cw1, C1T);
  transpose_to_f16<<<tg, tb, 0, stream>>>(cw2, C2T);
  prob_kernel<<<DIM / 256, 256, 0, stream>>>(uw1, prob1);
  prob_kernel<<<DIM / 256, 256, 0, stream>>>(uw2, prob2);

  dim3 gg(DIM / 128, DIM / 128);
  gemm_wmma<false, false, 0, true ><<<gg, 256, 0, stream>>>(x,  w0,  b0,  nullptr, Ha);
  gemm_wmma<true,  false, 0, true ><<<gg, 256, 0, stream>>>(Ha, w1,  b1,  nullptr, Hb);
  gemm_wmma<true,  true,  1, true ><<<gg, 256, 0, stream>>>(Hb, C1T, cb1, prob1,   Ha);
  gemm_wmma<true,  false, 0, true ><<<gg, 256, 0, stream>>>(Ha, w2,  b2,  nullptr, Hb);
  gemm_wmma<true,  true,  1, true ><<<gg, 256, 0, stream>>>(Hb, C2T, cb2, prob2,   Ha);
  gemm_wmma<true,  false, 0, false><<<gg, 256, 0, stream>>>(Ha, wf,  bf,  nullptr, out);
}
